// QuantizedGPT2MLP_90228672954734
// MI455X (gfx1250) — 8-bit matrix operand order repaired and checked by reading, compiled, not run on hardware
//
#include <hip/hip_runtime.h>
#include <hip/hip_bf16.h>
#include <stdint.h>

typedef __attribute__((ext_vector_type(8))) int  v8i;
typedef __attribute__((ext_vector_type(4))) unsigned int u32x4;
typedef __attribute__((ext_vector_type(8))) int  i32x8;
typedef __attribute__((ext_vector_type(4))) int  i32x4;

#if defined(__has_builtin)
#  if __has_builtin(__builtin_amdgcn_tensor_load_to_lds) && __has_builtin(__builtin_amdgcn_s_wait_tensorcnt)
#    define HAVE_TDM 1
#  endif
#endif
#ifndef HAVE_TDM
#  define HAVE_TDM 0
#endif

static constexpr size_t OFF_SCAL  = 0;
static constexpr size_t OFF_XQ    = 256;
static constexpr size_t OFF_WFC   = OFF_XQ   + 8192ull * 1024;
static constexpr size_t OFF_WPRJ  = OFF_WFC  + 4096ull * 1024;
static constexpr size_t OFF_HQ    = OFF_WPRJ + 4096ull * 1024;
static constexpr size_t OFF_T1    = OFF_HQ   + 8192ull * 4096;
static constexpr size_t OFF_T2    = OFF_T1   + 8192ull * 16 * 4;
static constexpr size_t OFF_H     = OFF_T2   + 8192ull * 16 * 4;


__global__ void init_scalars_kernel(int* s) {
    if (threadIdx.x < 8) s[threadIdx.x] = 0;
}

__global__ void absmax_kernel(const float4* __restrict__ x, size_t n4, int* __restrict__ slot) {
    __shared__ float red[256];
    float m = 0.0f;
    size_t stride = (size_t)gridDim.x * blockDim.x;
    for (size_t i = (size_t)blockIdx.x * blockDim.x + threadIdx.x; i < n4; i += stride) {
        float4 v = x[i];
        m = fmaxf(m, fmaxf(fmaxf(fabsf(v.x), fabsf(v.y)), fmaxf(fabsf(v.z), fabsf(v.w))));
    }
    red[threadIdx.x] = m;
    __syncthreads();
    for (int s = 128; s > 0; s >>= 1) {
        if ((int)threadIdx.x < s) red[threadIdx.x] = fmaxf(red[threadIdx.x], red[threadIdx.x + s]);
        __syncthreads();
    }
    if (threadIdx.x == 0) atomicMax(slot, __float_as_int(red[0]));
}

__global__ void quantize_kernel(const float* __restrict__ x, signed char* __restrict__ q,
                                size_t n4, const int* __restrict__ slot) {
    size_t i = (size_t)blockIdx.x * blockDim.x + threadIdx.x;
    if (i >= n4) return;
    float amax = __int_as_float(*slot);
    float inv  = 127.0f / fmaxf(amax, 1e-8f);
    float4 v = reinterpret_cast<const float4*>(x)[i];
    int a = (int)rintf(v.x * inv); a = max(-127, min(127, a));
    int b = (int)rintf(v.y * inv); b = max(-127, min(127, b));
    int c = (int)rintf(v.z * inv); c = max(-127, min(127, c));
    int d = (int)rintf(v.w * inv); d = max(-127, min(127, d));
    int packed = (a & 0xFF) | ((b & 0xFF) << 8) | ((c & 0xFF) << 16) | ((d & 0xFF) << 24);
    reinterpret_cast<int*>(q)[i] = packed;
}

#if HAVE_TDM
__device__ inline void tdm_load_2d(unsigned lds_off, const void* gaddr,
                                   unsigned tensor_d0, unsigned tensor_d1,
                                   unsigned tile_d0,   unsigned tile_d1,
                                   unsigned stride0_elems) {
    unsigned long long ga = (unsigned long long)(uintptr_t)gaddr;
    u32x4 g0;
    g0[0] = 1u;
    g0[1] = lds_off;
    g0[2] = (unsigned)(ga & 0xFFFFFFFFull);
    g0[3] = (unsigned)((ga >> 32) & 0x1FFFFFFull)
            | (2u << 30);
    i32x8 g1;
    g1[0] = (int)(2u << 16);
    g1[1] = (int)((tensor_d0 & 0xFFFFu) << 16);
    g1[2] = (int)(((tensor_d0 >> 16) & 0xFFFFu)
            | ((tensor_d1 & 0xFFFFu) << 16));
    g1[3] = (int)(((tensor_d1 >> 16) & 0xFFFFu)
            | ((tile_d0 & 0xFFFFu) << 16));
    g1[4] = (int)(tile_d1 & 0xFFFFu);
    g1[5] = (int)stride0_elems;
    g1[6] = 0;
    g1[7] = 0;
    i32x4 z4 = {0, 0, 0, 0};
    i32x8 z8 = {0, 0, 0, 0, 0, 0, 0, 0};
    __builtin_amdgcn_tensor_load_to_lds(g0, g1, z4, z4, z8, 0);
}
#endif

__global__ __launch_bounds__(256)
void lora_T_kernel(const float* __restrict__ X, const float* __restrict__ A,
                   float* __restrict__ T, int K) {
    __shared__ float xs[16][516];
    __shared__ float as_t[16][512];
    const int mi = threadIdx.x & 15;
    const int r  = threadIdx.x >> 4;
    const int m0 = blockIdx.x * 16;
    float acc = 0.0f;
#if HAVE_TDM
    const unsigned lds_off = (unsigned)(uintptr_t)(&as_t[0][0]);
#endif
    for (int k0 = 0; k0 < K; k0 += 512) {
#if HAVE_TDM
        if ((threadIdx.x >> 5) == 0) {
            tdm_load_2d(lds_off, A + k0, (unsigned)K, 16u, 512u, 16u, (unsigned)K);
        }
#else
        for (int t = threadIdx.x; t < 16 * 512; t += 256) {
            int row = t >> 9, kk = t & 511;
            as_t[row][kk] = A[(size_t)row * K + (k0 + kk)];
        }
#endif
        for (int t = threadIdx.x; t < 16 * 512; t += 256) {
            int row = t >> 9, kk = t & 511;
            xs[row][kk] = X[(size_t)(m0 + row) * K + (k0 + kk)];
        }
#if HAVE_TDM
        __builtin_amdgcn_s_wait_tensorcnt(0);
#endif
        __syncthreads();
        const float4* xv = reinterpret_cast<const float4*>(&xs[mi][0]);
        const float4* av = reinterpret_cast<const float4*>(&as_t[r][0]);
#pragma unroll 8
        for (int q = 0; q < 128; ++q) {
            float4 a4 = av[q];
            float4 x4 = xv[q];
            acc += x4.x * a4.x + x4.y * a4.y + x4.z * a4.z + x4.w * a4.w;
        }
        __syncthreads();
    }
    T[(size_t)(m0 + mi) * 16 + r] = acc;
}

__global__ __launch_bounds__(256)
void gemm_iu8_lora_kernel(const signed char* __restrict__ Aq,
                          const signed char* __restrict__ Bq,
                          const float* __restrict__ bias,
                          const float* __restrict__ T,
                          const float* __restrict__ BL,
                          const float* __restrict__ amaxA,
                          const float* __restrict__ amaxB,
                          float* __restrict__ out,
                          int M, int N, int K, int apply_gelu,
                          int* __restrict__ amax_out) {
    __shared__ float red[256];
    const int lane  = threadIdx.x & 31;
    const int wave  = threadIdx.x >> 5;
    const int waveM = wave >> 2;
    const int waveN = wave & 3;
    const int m_base = blockIdx.y * 128 + waveM * 64;
    const int n_base = blockIdx.x * 128 + waveN * 32;

    const int half = lane >> 4;
    const int lid  = lane & 15;

    v8i zero = {};
    v8i acc[4][2];
#pragma unroll
    for (int i = 0; i < 4; ++i)
#pragma unroll
        for (int j = 0; j < 2; ++j) acc[i][j] = zero;

    int offA[4];
#pragma unroll
    for (int i = 0; i < 4; ++i)
        offA[i] = (m_base + i * 16 + lid) * K + half * 16;
    int offB[2];
#pragma unroll
    for (int j = 0; j < 2; ++j)
        offB[j] = (n_base + j * 16 + lid) * K + half * 16;

#pragma unroll
    for (int i = 0; i < 4; ++i) __builtin_prefetch(Aq + offA[i], 0, 3);
#pragma unroll
    for (int j = 0; j < 2; ++j) __builtin_prefetch(Bq + offB[j], 0, 3);

    for (int k0 = 0; k0 < K; k0 += 64) {
        const signed char* baseA = Aq + k0;
        const signed char* baseB = Bq + k0;
        v8i a[4], b[2];
#pragma unroll
        for (int i = 0; i < 4; ++i) {
            const int2* q = reinterpret_cast<const int2*>(baseA + offA[i]);
            int2 x0 = q[0], x1 = q[1], x2 = q[4], x3 = q[5];
            a[i][0] = x0.x; a[i][1] = x0.y; a[i][2] = x1.x; a[i][3] = x1.y;
            a[i][4] = x2.x; a[i][5] = x2.y; a[i][6] = x3.x; a[i][7] = x3.y;
        }
#pragma unroll
        for (int j = 0; j < 2; ++j) {
            const int4* q = reinterpret_cast<const int4*>(baseB + offB[j]);
            int4 y0 = q[0], y1 = q[2];
            b[j][0] = y0.x; b[j][1] = y0.y; b[j][2] = y0.z; b[j][3] = y0.w;
            b[j][4] = y1.x; b[j][5] = y1.y; b[j][6] = y1.z; b[j][7] = y1.w;
        }
#pragma unroll
        for (int i = 0; i < 4; ++i)
#pragma unroll
            for (int j = 0; j < 2; ++j)
                acc[i][j] = __builtin_amdgcn_wmma_i32_16x16x64_iu8(
                    true, a[i], true, b[j], acc[i][j], false, false);
    }

    const float sA = fmaxf(amaxA[0], 1e-8f) / 127.0f;
    const float sB = fmaxf(amaxB[0], 1e-8f) / 127.0f;
    const float s  = sA * sB;

    int    ncol[2];
    float  bn[2];
    float4 bl[2][4];
#pragma unroll
    for (int j = 0; j < 2; ++j) {
        ncol[j] = n_base + j * 16 + lid;
        bn[j]   = bias[ncol[j]];
        const float4* p = reinterpret_cast<const float4*>(BL + (size_t)ncol[j] * 16);
        bl[j][0] = p[0]; bl[j][1] = p[1]; bl[j][2] = p[2]; bl[j][3] = p[3];
    }

    float vmax = 0.0f;
#pragma unroll
    for (int i = 0; i < 4; ++i) {
#pragma unroll
        for (int g = 0; g < 8; ++g) {
            int m = m_base + i * 16 + g + half * 8;
            const float4* tr = reinterpret_cast<const float4*>(T + (size_t)m * 16);
            float4 t0 = tr[0], t1 = tr[1], t2 = tr[2], t3 = tr[3];
#pragma unroll
            for (int j = 0; j < 2; ++j) {
                float lora =
                    t0.x * bl[j][0].x + t0.y * bl[j][0].y + t0.z * bl[j][0].z + t0.w * bl[j][0].w +
                    t1.x * bl[j][1].x + t1.y * bl[j][1].y + t1.z * bl[j][1].z + t1.w * bl[j][1].w +
                    t2.x * bl[j][2].x + t2.y * bl[j][2].y + t2.z * bl[j][2].z + t2.w * bl[j][2].w +
                    t3.x * bl[j][3].x + t3.y * bl[j][3].y + t3.z * bl[j][3].z + t3.w * bl[j][3].w;
                float v = (float)acc[i][j][g] * s + bn[j] + 2.0f * lora;
                if (apply_gelu) v = 0.5f * v * (1.0f + erff(v * 0.70710678118654752f));
                vmax = fmaxf(vmax, fabsf(v));
                out[(size_t)m * N + ncol[j]] = v;
            }
        }
    }

    if (amax_out) {
        red[threadIdx.x] = vmax;
        __syncthreads();
        for (int s2 = 128; s2 > 0; s2 >>= 1) {
            if ((int)threadIdx.x < s2)
                red[threadIdx.x] = fmaxf(red[threadIdx.x], red[threadIdx.x + s2]);
            __syncthreads();
        }
        if (threadIdx.x == 0) atomicMax(amax_out, __float_as_int(red[0]));
    }
}


extern "C" void kernel_launch(void* const* d_in, const int* in_sizes, int n_in,
                              void* d_out, int out_size, void* d_ws, size_t ws_size,
                              hipStream_t stream) {
    (void)in_sizes; (void)n_in; (void)out_size; (void)ws_size;

    const float* x      = (const float*)d_in[0];
    const float* W_fc   = (const float*)d_in[1];
    const float* b_fc   = (const float*)d_in[2];
    const float* A_fc   = (const float*)d_in[3];
    const float* B_fc   = (const float*)d_in[4];
    const float* W_prj  = (const float*)d_in[5];
    const float* b_prj  = (const float*)d_in[6];
    const float* A_prj  = (const float*)d_in[7];
    const float* B_prj  = (const float*)d_in[8];
    float* out = (float*)d_out;

    char* ws = (char*)d_ws;
    int*         slots = (int*)(ws + OFF_SCAL);
    signed char* xq    = (signed char*)(ws + OFF_XQ);
    signed char* wfc8  = (signed char*)(ws + OFF_WFC);
    signed char* wprj8 = (signed char*)(ws + OFF_WPRJ);
    signed char* hq    = (signed char*)(ws + OFF_HQ);
    float*       T1    = (float*)(ws + OFF_T1);
    float*       T2    = (float*)(ws + OFF_T2);
    float*       h     = (float*)(ws + OFF_H);

    const int M = 8192, Dff = 4096, Dm = 1024;

    init_scalars_kernel<<<1, 32, 0, stream>>>(slots);

    absmax_kernel<<<1024, 256, 0, stream>>>((const float4*)x,     (size_t)M * Dm / 4,   slots + 0);
    absmax_kernel<<< 512, 256, 0, stream>>>((const float4*)W_fc,  (size_t)Dff * Dm / 4, slots + 1);
    absmax_kernel<<< 512, 256, 0, stream>>>((const float4*)W_prj, (size_t)Dm * Dff / 4, slots + 2);

    {
        size_t n4 = (size_t)M * Dm / 4;
        quantize_kernel<<<(unsigned)((n4 + 255) / 256), 256, 0, stream>>>(x, xq, n4, slots + 0);
    }
    {
        size_t n4 = (size_t)Dff * Dm / 4;
        quantize_kernel<<<(unsigned)((n4 + 255) / 256), 256, 0, stream>>>(W_fc, wfc8, n4, slots + 1);
    }
    {
        size_t n4 = (size_t)Dm * Dff / 4;
        quantize_kernel<<<(unsigned)((n4 + 255) / 256), 256, 0, stream>>>(W_prj, wprj8, n4, slots + 2);
    }

    lora_T_kernel<<<M / 16, 256, 0, stream>>>(x, A_fc, T1, Dm);

    {
        dim3 grid(Dff / 128, M / 128);
        gemm_iu8_lora_kernel<<<grid, 256, 0, stream>>>(
            xq, wfc8, b_fc, T1, B_fc,
            (const float*)(slots + 0), (const float*)(slots + 1),
            h, M, Dff, Dm,  1,  slots + 3);
    }

    {
        size_t n4 = (size_t)M * Dff / 4;
        quantize_kernel<<<(unsigned)((n4 + 255) / 256), 256, 0, stream>>>(h, hq, n4, slots + 3);
    }

    lora_T_kernel<<<M / 16, 256, 0, stream>>>(h, A_prj, T2, Dff);

    {
        dim3 grid(Dm / 128, M / 128);
        gemm_iu8_lora_kernel<<<grid, 256, 0, stream>>>(
            hq, wprj8, b_prj, T2, B_prj,
            (const float*)(slots + 3), (const float*)(slots + 2),
            out, M, Dm, Dff,  0,  nullptr);
    }
}
